// TransformerBlock_29033978921051
// MI455X (gfx1250) — matrix operand order repaired and checked by reading, not compiled, not run on hardware
//
#include <hip/hip_runtime.h>
#include <hip/hip_bf16.h>
#include <math.h>

#define DIM    192
#define HEADS  6
#define CP     32
#define HIDDEN 384
#define BATCH  4
#define HW     16384
#define NPIX   (BATCH * HW)   // 65536
#define C3     (3 * DIM)      // 576
#define KCHUNKS 16            // K-split for q.k^T

typedef __attribute__((ext_vector_type(16))) __bf16 v16bf;
typedef __attribute__((ext_vector_type(8)))  float  v8f;
typedef unsigned int u32x4 __attribute__((ext_vector_type(4)));
typedef int          i32x4 __attribute__((ext_vector_type(4)));
typedef int          i32x8 __attribute__((ext_vector_type(8)));

// ---------------- workspace layout (bytes) ----------------
constexpr size_t S_X1F   = (size_t)BATCH * DIM * HW * 4;      // 50331648
constexpr size_t S_R     = (size_t)BATCH * C3 * HW * 4;       // 150994944
constexpr size_t S_QKVB  = (size_t)BATCH * C3 * HW * 2;       // 75497472
constexpr size_t S_X1B   = (size_t)NPIX * DIM * 2;            // 25165824
constexpr size_t S_ATT   = (size_t)BATCH * HEADS * CP * CP * 4; // 98304

constexpr size_t OFF_X1F   = 0;
constexpr size_t OFF_R1    = OFF_X1F + S_X1F;                 // qkv_pre f32
constexpr size_t OFF_X2    = OFF_R1;                          // reuse: x2 pixel-major f32
constexpr size_t OFF_YBF   = OFF_R1 + S_X1F;                  // reuse: ln2 out bf16
constexpr size_t OFF_H1    = OFF_YBF + S_X1B;                 // reuse: fc1 out bf16 [N][384]
constexpr size_t OFF_R2    = OFF_R1 + S_R;                    // qkv after dwconv f32
constexpr size_t OFF_OUTB  = OFF_R2;                          // reuse: attn@v out bf16 pixel-major
constexpr size_t OFF_Y4    = OFF_R2 + S_X1B;                  // reuse: y4 f32 channel-major
constexpr size_t OFF_GATE  = OFF_Y4 + S_X1F;                  // reuse: gate f32 [B][HW]
constexpr size_t OFF_QKVB  = OFF_R2 + S_R;                    // q/k/v bf16 channel-major
constexpr size_t OFF_X1B   = OFF_QKVB + S_QKVB;               // x1 bf16 pixel-major
constexpr size_t OFF_ATNR  = OFF_X1B + S_X1B;                 // attn raw f32 [4][6][32][32]
constexpr size_t OFF_ATNC  = OFF_ATNR + S_ATT;                // attn after head-conv f32
constexpr size_t OFF_ATNB  = OFF_ATNC + S_ATT;                // attn softmax bf16
constexpr size_t OFF_WQKV  = OFF_ATNB + S_ATT / 2;            // bf16 weights
constexpr size_t OFF_WPROJ = OFF_WQKV + 576 * 192 * 2;
constexpr size_t OFF_WFC1  = OFF_WPROJ + 192 * 192 * 2;
constexpr size_t OFF_WFC2  = OFF_WFC1 + 384 * 192 * 2;
constexpr size_t OFF_ATNP  = OFF_WFC2 + 192 * 384 * 2;        // attn partials [16][4][6][32][32] f32
constexpr size_t WS_NEEDED = OFF_ATNP + (size_t)KCHUNKS * S_ATT;  // ~435 MB

// ---------------- helpers ----------------
__device__ __forceinline__ float warp_sum(float v) {
#pragma unroll
  for (int m = 16; m > 0; m >>= 1) v += __shfl_xor(v, m, 32);
  return v;
}

__device__ __forceinline__ float gelu_exact(float x) {
  return 0.5f * x * (1.0f + erff(x * 0.7071067811865476f));
}

// One 16x16 tile per wave. A:[M][K] row-major (lda), Bt:[N][K] row-major (ldb).
// A/B use contiguous k = 16*hf + i (same in both); C/D per CDNA5 ISA 7.12.2 (16-bit A 16x32, B 32x16, f32 C/D 16x16).
__device__ __forceinline__ v8f wmma_gemm_rowrow(const __bf16* __restrict__ A, size_t lda,
                                                const __bf16* __restrict__ Bt, size_t ldb,
                                                int K) {
  const int lane = threadIdx.x & 31;
  const int hf   = lane >> 4;
  const int lr   = lane & 15;
  const __bf16* ap = A  + (size_t)lr * lda;
  const __bf16* bp = Bt + (size_t)lr * ldb;
  v8f acc = {};
  for (int k0 = 0; k0 < K; k0 += 32) {
    if (K >= 1024 && k0 + 256 < K) {             // long-K streaming path
      __builtin_prefetch(ap + k0 + 256, 0, 0);   // -> global_prefetch_b8
      __builtin_prefetch(bp + k0 + 256, 0, 0);
    }
    v16bf a, b;
#pragma unroll
    for (int i = 0; i < 16; ++i) a[i]    = ap[k0 + hf * 16 + i];
#pragma unroll
    for (int i = 0; i < 16; ++i) b[i]    = bp[k0 + hf * 16 + i];
    acc = __builtin_amdgcn_wmma_f32_16x16x32_bf16(false, a, false, b, (short)0, acc, false, false);
  }
  return acc;
}

// 16x32 tile per wave: one A fragment feeds two WMMAs (halves A traffic).
__device__ __forceinline__ void wmma_gemm_rowrow_x2(const __bf16* __restrict__ A, size_t lda,
                                                    const __bf16* __restrict__ Bt0,
                                                    const __bf16* __restrict__ Bt1, size_t ldb,
                                                    int K, v8f& acc0, v8f& acc1) {
  const int lane = threadIdx.x & 31;
  const int hf   = lane >> 4;
  const int lr   = lane & 15;
  const __bf16* ap  = A   + (size_t)lr * lda;
  const __bf16* bp0 = Bt0 + (size_t)lr * ldb;
  const __bf16* bp1 = Bt1 + (size_t)lr * ldb;
  for (int k0 = 0; k0 < K; k0 += 32) {
    v16bf a, b0, b1;
#pragma unroll
    for (int i = 0; i < 16; ++i) a[i]    = ap[k0 + hf * 16 + i];
#pragma unroll
    for (int i = 0; i < 16; ++i) b0[i]   = bp0[k0 + hf * 16 + i];
#pragma unroll
    for (int i = 0; i < 16; ++i) b1[i]   = bp1[k0 + hf * 16 + i];
    acc0 = __builtin_amdgcn_wmma_f32_16x16x32_bf16(false, a, false, b0, (short)0, acc0, false, false);
    acc1 = __builtin_amdgcn_wmma_f32_16x16x32_bf16(false, a, false, b1, (short)0, acc1, false, false);
  }
}

// ---------------- TDM probe: NULL descriptor (count=0) => architectural no-op ----------------
#if defined(__has_builtin)
#if __has_builtin(__builtin_amdgcn_tensor_load_to_lds)
#define HAVE_TDM 1
#endif
#endif

__global__ void tdm_probe_kernel() {
#ifdef HAVE_TDM
  u32x4 g0 = {};   // count[1:0]=0 -> NULL tensor, no data moved
  i32x8 g1 = {};
  i32x4 g2 = {};
  i32x4 g3 = {};
#if __clang_major__ >= 23
  i32x8 g4 = {};
  __builtin_amdgcn_tensor_load_to_lds(g0, g1, g2, g3, g4, 0);
#else
  __builtin_amdgcn_tensor_load_to_lds(g0, g1, g2, g3, 0);
#endif
  __builtin_amdgcn_s_wait_tensorcnt(0);
#endif
}

// ---------------- kernels ----------------
__global__ void cvt_bf16_kernel(const float* __restrict__ src, __bf16* __restrict__ dst, int n) {
  int i = blockIdx.x * blockDim.x + threadIdx.x;
  if (i < n) dst[i] = (__bf16)src[i];
}

// LayerNorm over C=192 per pixel; block = 192 threads, one pixel per block.
__global__ void ln1_kernel(const float* __restrict__ x, const float* __restrict__ g,
                           const float* __restrict__ bta, float* __restrict__ x1f,
                           __bf16* __restrict__ x1b) {
  const int n = blockIdx.x;
  const int c = threadIdx.x;
  const int b = n >> 14, p = n & (HW - 1);
  float v  = x[((size_t)(b * DIM + c)) * HW + p];
  float s  = warp_sum(v);
  float s2 = warp_sum(v * v);
  __shared__ float ls[6], ls2[6];
  __shared__ float mean_s, inv_s;
  const int wid = threadIdx.x >> 5, lane = threadIdx.x & 31;
  if (lane == 0) { ls[wid] = s; ls2[wid] = s2; }
  __syncthreads();
  if (threadIdx.x == 0) {
    float t = 0.f, t2 = 0.f;
#pragma unroll
    for (int i = 0; i < 6; ++i) { t += ls[i]; t2 += ls2[i]; }
    float m   = t / DIM;
    float var = t2 / DIM - m * m;
    mean_s = m;
    inv_s  = rsqrtf(var + 1e-5f);
  }
  __syncthreads();
  float o = (v - mean_s) * inv_s * g[c] + bta[c];
  x1f[((size_t)(b * DIM + c)) * HW + p] = o;
  x1b[(size_t)n * DIM + c] = (__bf16)o;
}

// qkv 1x1: [N x 192] * wqkv[576 x 192]^T -> pre[b][oc][p]; 16x32 tile per wave
__global__ void qkv_gemm_kernel(const __bf16* __restrict__ x1b, const __bf16* __restrict__ w,
                                float* __restrict__ out) {
  const int wid  = (blockIdx.x * blockDim.x + threadIdx.x) >> 5;
  const int lane = threadIdx.x & 31, hf = lane >> 4, lr = lane & 15;
  const int ocTiles = C3 / 32;                          // 18
  const int pixTile = wid / ocTiles, ocTile = wid % ocTiles;
  const int n0 = pixTile * 16, oc0 = ocTile * 32;
  v8f acc0 = {}, acc1 = {};
  wmma_gemm_rowrow_x2(x1b + (size_t)n0 * DIM, DIM,
                      w + (size_t)oc0 * DIM, w + (size_t)(oc0 + 16) * DIM, DIM, DIM, acc0, acc1);
  const int b = n0 >> 14, p0 = n0 & (HW - 1);
  float* op0 = out + ((size_t)(b * C3 + oc0 + lr)) * HW + p0 + hf * 8;
  float* op1 = out + ((size_t)(b * C3 + oc0 + 16 + lr)) * HW + p0 + hf * 8;
#pragma unroll
  for (int v = 0; v < 8; ++v) { op0[v] = acc0[v]; op1[v] = acc1[v]; }
}

// depthwise 3x3, pad 1, groups = 576
__global__ void dwconv_kernel(const float* __restrict__ in, const float* __restrict__ wdw,
                              float* __restrict__ out) {
  int idx = blockIdx.x * blockDim.x + threadIdx.x;
  if (idx >= BATCH * C3 * HW) return;
  const int p = idx & (HW - 1);
  const int c = (idx >> 14) % C3;
  const int b = idx / (C3 * HW);
  const int y = p >> 7, x = p & 127;
  const float* base = in + ((size_t)(b * C3 + c)) * HW;
  const float* wp   = wdw + c * 9;
  float acc = 0.f;
#pragma unroll
  for (int dy = -1; dy <= 1; ++dy) {
    int yy = y + dy;
    if (yy < 0 || yy > 127) continue;
#pragma unroll
    for (int dx = -1; dx <= 1; ++dx) {
      int xx = x + dx;
      if (xx < 0 || xx > 127) continue;
      acc += base[yy * 128 + xx] * wp[(dy + 1) * 3 + (dx + 1)];
    }
  }
  out[idx] = acc;
}

// per (b, channel): L2-normalize q,k over hw; convert q,k,v to bf16
__global__ void qk_norm_cvt_kernel(const float* __restrict__ qkv, __bf16* __restrict__ qkvb) {
  const int bc = blockIdx.x;
  const int c  = bc % C3;
  const float* src = qkv + (size_t)bc * HW;
  __bf16* dst = qkvb + (size_t)bc * HW;
  __shared__ float ls[8];
  __shared__ float sc;
  float scale = 1.0f;
  if (c < 2 * DIM) {
    float ss = 0.f;
    for (int i = threadIdx.x; i < HW; i += blockDim.x) { float v = src[i]; ss += v * v; }
    ss = warp_sum(ss);
    const int wid = threadIdx.x >> 5, lane = threadIdx.x & 31;
    if (lane == 0) ls[wid] = ss;
    __syncthreads();
    if (threadIdx.x == 0) {
      float t = 0.f;
      for (int i = 0; i < (int)(blockDim.x >> 5); ++i) t += ls[i];
      sc = 1.0f / fmaxf(sqrtf(t), 1e-12f);
    }
    __syncthreads();
    scale = sc;
  }
  for (int i = threadIdx.x; i < HW; i += blockDim.x) dst[i] = (__bf16)(src[i] * scale);
}

// q.k^T partials: K split into 16 chunks of 1024 -> 1536 waves (deterministic, no atomics)
__global__ void attn_qk_part_kernel(const __bf16* __restrict__ qkvb, float* __restrict__ part) {
  const int wid  = (blockIdx.x * blockDim.x + threadIdx.x) >> 5;
  const int lane = threadIdx.x & 31, hf = lane >> 4, lr = lane & 15;
  const int chunk = wid & (KCHUNKS - 1);
  const int t  = (wid >> 4) & 3;
  const int bh = wid >> 6;
  const int ti = t >> 1, tj = t & 1;
  const int b = bh / HEADS, hd = bh % HEADS;
  const int n0 = chunk * (HW / KCHUNKS);
  const __bf16* A  = qkvb + ((size_t)(b * C3 + hd * CP + ti * 16)) * HW + n0;
  const __bf16* Bt = qkvb + ((size_t)(b * C3 + DIM + hd * CP + tj * 16)) * HW + n0;
  v8f acc = wmma_gemm_rowrow(A, HW, Bt, HW, HW / KCHUNKS);
  float* op = part + (size_t)chunk * (BATCH * HEADS * CP * CP)
                   + (((size_t)bh * CP + ti * 16) * CP) + tj * 16;
#pragma unroll
  for (int v = 0; v < 8; ++v) op[(v + 8 * hf) * CP + lr] = acc[v];
}

// sum partials + apply temp (reference scales before head-conv)
__global__ void attn_reduce_kernel(const float* __restrict__ part, const float* __restrict__ temp,
                                   float* __restrict__ attn) {
  int idx = blockIdx.x * blockDim.x + threadIdx.x;
  if (idx >= BATCH * HEADS * CP * CP) return;
  const int hd = (idx >> 10) % HEADS;
  float s = 0.f;
#pragma unroll
  for (int c = 0; c < KCHUNKS; ++c) s += part[(size_t)c * (BATCH * HEADS * CP * CP) + idx];
  attn[idx] = s * temp[hd];
}

// 3x3 conv over head channels on attn [4][6][32][32]
__global__ void attn_head_conv_kernel(const float* __restrict__ a, const float* __restrict__ wh,
                                      float* __restrict__ out) {
  int idx = blockIdx.x * blockDim.x + threadIdx.x;
  if (idx >= BATCH * HEADS * CP * CP) return;
  const int j = idx & 31, i = (idx >> 5) & 31;
  const int ho = (idx >> 10) % HEADS, b = (idx >> 10) / HEADS;
  float acc = 0.f;
  for (int hi = 0; hi < HEADS; ++hi) {
    const float* ap = a + (size_t)(b * HEADS + hi) * CP * CP;
    const float* wp = wh + (ho * HEADS + hi) * 9;
#pragma unroll
    for (int dy = -1; dy <= 1; ++dy) {
      int ii = i + dy;
      if (ii < 0 || ii > 31) continue;
#pragma unroll
      for (int dx = -1; dx <= 1; ++dx) {
        int jj = j + dx;
        if (jj < 0 || jj > 31) continue;
        acc += ap[ii * CP + jj] * wp[(dy + 1) * 3 + (dx + 1)];
      }
    }
  }
  out[idx] = acc;
}

// softmax over last dim (32); one wave per row; outputs bf16 for WMMA
__global__ void softmax_kernel(const float* __restrict__ a, __bf16* __restrict__ out) {
  const int row = blockIdx.x;
  const int lane = threadIdx.x;
  float v = a[(size_t)row * CP + lane];
  float mx = v;
#pragma unroll
  for (int m = 16; m > 0; m >>= 1) mx = fmaxf(mx, __shfl_xor(mx, m, 32));
  float e = expf(v - mx);
  float s = warp_sum(e);
  out[(size_t)row * CP + lane] = (__bf16)(e / s);
}

// out[c][n] = sum_d attn[c][d] v[d][n]; single WMMA per 16x16 tile (K = 32)
__global__ void attn_v_kernel(const __bf16* __restrict__ attnb, const __bf16* __restrict__ qkvb,
                              __bf16* __restrict__ outb) {
  const int wid  = (blockIdx.x * blockDim.x + threadIdx.x) >> 5;
  const int lane = threadIdx.x & 31, hf = lane >> 4, lr = lane & 15;
  const int nTile = wid & 1023;
  const int rest  = wid >> 10;
  const int ti = rest & 1;
  const int bh = rest >> 1;
  const int b = bh / HEADS, hd = bh % HEADS;
  const int n0 = nTile * 16;
  const __bf16* ap = attnb + ((size_t)((b * HEADS + hd) * CP + ti * 16 + lr)) * CP;
  v16bf a, bb;
#pragma unroll
  for (int i = 0; i < 16; ++i) a[i] = ap[hf * 16 + i];
  const __bf16* vp = qkvb + ((size_t)(b * C3 + 2 * DIM + hd * CP)) * HW + n0 + lr;
#pragma unroll
  for (int i = 0; i < 16; ++i) bb[i] = vp[(size_t)(hf * 16 + i) * HW];
  v8f acc = {};
  acc = __builtin_amdgcn_wmma_f32_16x16x32_bf16(false, a, false, bb, (short)0, acc, false, false);
  __bf16* op = outb + ((size_t)(b * HW + n0 + lr)) * DIM + hd * CP + ti * 16 + hf * 8;
#pragma unroll
  for (int v = 0; v < 8; ++v) op[v] = (__bf16)acc[v];
}

// proj 1x1 + residual: x2[n][oc] = x1[b][oc][p] + sum_ic out[n][ic] wproj[oc][ic]
__global__ void proj_gemm_kernel(const __bf16* __restrict__ outb, const __bf16* __restrict__ w,
                                 const float* __restrict__ x1f, float* __restrict__ x2) {
  const int wid  = (blockIdx.x * blockDim.x + threadIdx.x) >> 5;
  const int lane = threadIdx.x & 31, hf = lane >> 4, lr = lane & 15;
  const int ocTiles = DIM / 32;                         // 6
  const int pixTile = wid / ocTiles, ocTile = wid % ocTiles;
  const int n0 = pixTile * 16, oc0 = ocTile * 32;
  v8f acc0 = {}, acc1 = {};
  wmma_gemm_rowrow_x2(outb + (size_t)n0 * DIM, DIM,
                      w + (size_t)oc0 * DIM, w + (size_t)(oc0 + 16) * DIM, DIM, DIM, acc0, acc1);
  const int b = n0 >> 14, p0 = n0 & (HW - 1);
  const int oc = oc0 + lr;
  const float* xp0 = x1f + ((size_t)(b * DIM + oc)) * HW + p0 + hf * 8;
  const float* xp1 = x1f + ((size_t)(b * DIM + oc + 16)) * HW + p0 + hf * 8;
  float* op0 = x2 + ((size_t)(n0 + hf * 8)) * DIM + oc;
  float* op1 = op0 + 16;
#pragma unroll
  for (int v = 0; v < 8; ++v) {
    op0[(size_t)v * DIM] = xp0[v] + acc0[v];
    op1[(size_t)v * DIM] = xp1[v] + acc1[v];
  }
}

// LayerNorm2 on pixel-major x2; emit bf16 pixel-major for fc1
__global__ void ln2_kernel(const float* __restrict__ x2, const float* __restrict__ g,
                           const float* __restrict__ bta, __bf16* __restrict__ ybf) {
  const int n = blockIdx.x;
  const int c = threadIdx.x;
  float v  = x2[(size_t)n * DIM + c];
  float s  = warp_sum(v);
  float s2 = warp_sum(v * v);
  __shared__ float ls[6], ls2[6];
  __shared__ float mean_s, inv_s;
  const int wid = threadIdx.x >> 5, lane = threadIdx.x & 31;
  if (lane == 0) { ls[wid] = s; ls2[wid] = s2; }
  __syncthreads();
  if (threadIdx.x == 0) {
    float t = 0.f, t2 = 0.f;
#pragma unroll
    for (int i = 0; i < 6; ++i) { t += ls[i]; t2 += ls2[i]; }
    float m   = t / DIM;
    float var = t2 / DIM - m * m;
    mean_s = m;
    inv_s  = rsqrtf(var + 1e-5f);
  }
  __syncthreads();
  float o = (v - mean_s) * inv_s * g[c] + bta[c];
  ybf[(size_t)n * DIM + c] = (__bf16)o;
}

// fc1: [N x 192] * wfc1[384 x 192]^T, + bias, GELU, store bf16 pixel-major
__global__ void fc1_gemm_kernel(const __bf16* __restrict__ ybf, const __bf16* __restrict__ w,
                                const float* __restrict__ bias, __bf16* __restrict__ h1) {
  const int wid  = (blockIdx.x * blockDim.x + threadIdx.x) >> 5;
  const int lane = threadIdx.x & 31, hf = lane >> 4, lr = lane & 15;
  const int ocTiles = HIDDEN / 32;                      // 12
  const int pixTile = wid / ocTiles, ocTile = wid % ocTiles;
  const int n0 = pixTile * 16, oc0 = ocTile * 32;
  v8f acc0 = {}, acc1 = {};
  wmma_gemm_rowrow_x2(ybf + (size_t)n0 * DIM, DIM,
                      w + (size_t)oc0 * DIM, w + (size_t)(oc0 + 16) * DIM, DIM, DIM, acc0, acc1);
  const int oc = oc0 + lr;
  const float bv0 = bias[oc], bv1 = bias[oc + 16];
  __bf16* op0 = h1 + ((size_t)(n0 + hf * 8)) * HIDDEN + oc;
  __bf16* op1 = op0 + 16;
#pragma unroll
  for (int v = 0; v < 8; ++v) {
    op0[(size_t)v * HIDDEN] = (__bf16)gelu_exact(acc0[v] + bv0);
    op1[(size_t)v * HIDDEN] = (__bf16)gelu_exact(acc1[v] + bv1);
  }
}

// fc2: [N x 384] * wfc2[192 x 384]^T, + bias, GELU, store f32 channel-major y4
__global__ void fc2_gemm_kernel(const __bf16* __restrict__ h1, const __bf16* __restrict__ w,
                                const float* __restrict__ bias, float* __restrict__ y4) {
  const int wid  = (blockIdx.x * blockDim.x + threadIdx.x) >> 5;
  const int lane = threadIdx.x & 31, hf = lane >> 4, lr = lane & 15;
  const int ocTiles = DIM / 32;                         // 6
  const int pixTile = wid / ocTiles, ocTile = wid % ocTiles;
  const int n0 = pixTile * 16, oc0 = ocTile * 32;
  v8f acc0 = {}, acc1 = {};
  wmma_gemm_rowrow_x2(h1 + (size_t)n0 * HIDDEN, HIDDEN,
                      w + (size_t)oc0 * HIDDEN, w + (size_t)(oc0 + 16) * HIDDEN, HIDDEN, HIDDEN,
                      acc0, acc1);
  const int b = n0 >> 14, p0 = n0 & (HW - 1);
  const int oc = oc0 + lr;
  const float bv0 = bias[oc], bv1 = bias[oc + 16];
  float* op0 = y4 + ((size_t)(b * DIM + oc)) * HW + p0 + hf * 8;
  float* op1 = y4 + ((size_t)(b * DIM + oc + 16)) * HW + p0 + hf * 8;
#pragma unroll
  for (int v = 0; v < 8; ++v) {
    op0[v] = gelu_exact(acc0[v] + bv0);
    op1[v] = gelu_exact(acc1[v] + bv1);
  }
}

// gate = sigmoid(conv3x3(y4, w_sa) + b_sa) : one thread per pixel
__global__ void gate_kernel(const float* __restrict__ y4, const float* __restrict__ wsa,
                            const float* __restrict__ bsa, float* __restrict__ gate) {
  int idx = blockIdx.x * blockDim.x + threadIdx.x;
  if (idx >= BATCH * HW) return;
  const int p = idx & (HW - 1), b = idx >> 14;
  const int y = p >> 7, x = p & 127;
  float acc = bsa[0];
  for (int c = 0; c < DIM; ++c) {
    const float* base = y4 + ((size_t)(b * DIM + c)) * HW;
    const float* wp   = wsa + c * 9;
#pragma unroll
    for (int dy = -1; dy <= 1; ++dy) {
      int yy = y + dy;
      if (yy < 0 || yy > 127) continue;
#pragma unroll
      for (int dx = -1; dx <= 1; ++dx) {
        int xx = x + dx;
        if (xx < 0 || xx > 127) continue;
        acc += base[yy * 128 + xx] * wp[(dy + 1) * 3 + (dx + 1)];
      }
    }
  }
  gate[idx] = 1.0f / (1.0f + expf(-acc));
}

// out[b][c][p] = x2[n][c] + y4[b][c][p] * gate[b][p]
__global__ void final_kernel(const float* __restrict__ x2, const float* __restrict__ y4,
                             const float* __restrict__ gate, float* __restrict__ out) {
  int idx = blockIdx.x * blockDim.x + threadIdx.x;
  if (idx >= BATCH * DIM * HW) return;
  const int p = idx & (HW - 1);
  const int c = (idx >> 14) % DIM;
  const int b = idx / (DIM * HW);
  out[idx] = x2[((size_t)(b * HW + p)) * DIM + c] + y4[idx] * gate[b * HW + p];
}

// ---------------- launcher ----------------
extern "C" void kernel_launch(void* const* d_in, const int* in_sizes, int n_in,
                              void* d_out, int out_size, void* d_ws, size_t ws_size,
                              hipStream_t stream) {
  (void)in_sizes; (void)n_in; (void)out_size; (void)ws_size;
  const float* x      = (const float*)d_in[0];
  const float* g1     = (const float*)d_in[1];
  const float* b1     = (const float*)d_in[2];
  const float* w_qkv  = (const float*)d_in[3];
  const float* w_dw   = (const float*)d_in[4];
  const float* temp   = (const float*)d_in[5];
  const float* w_head = (const float*)d_in[6];
  const float* w_proj = (const float*)d_in[7];
  const float* g2     = (const float*)d_in[8];
  const float* b2     = (const float*)d_in[9];
  const float* w_fc1  = (const float*)d_in[10];
  const float* b_fc1  = (const float*)d_in[11];
  const float* w_fc2  = (const float*)d_in[12];
  const float* b_fc2  = (const float*)d_in[13];
  const float* w_sa   = (const float*)d_in[14];
  const float* b_sa   = (const float*)d_in[15];
  float* out = (float*)d_out;

  char* ws = (char*)d_ws;
  float*  x1f   = (float*) (ws + OFF_X1F);
  float*  qkvp  = (float*) (ws + OFF_R1);
  float*  x2    = (float*) (ws + OFF_X2);
  __bf16* ybf   = (__bf16*)(ws + OFF_YBF);
  __bf16* h1    = (__bf16*)(ws + OFF_H1);
  float*  qkvf  = (float*) (ws + OFF_R2);
  __bf16* outb  = (__bf16*)(ws + OFF_OUTB);
  float*  y4    = (float*) (ws + OFF_Y4);
  float*  gate  = (float*) (ws + OFF_GATE);
  __bf16* qkvb  = (__bf16*)(ws + OFF_QKVB);
  __bf16* x1b   = (__bf16*)(ws + OFF_X1B);
  float*  atnr  = (float*) (ws + OFF_ATNR);
  float*  atnc  = (float*) (ws + OFF_ATNC);
  __bf16* atnb  = (__bf16*)(ws + OFF_ATNB);
  __bf16* wqkvb = (__bf16*)(ws + OFF_WQKV);
  __bf16* wprjb = (__bf16*)(ws + OFF_WPROJ);
  __bf16* wfc1b = (__bf16*)(ws + OFF_WFC1);
  __bf16* wfc2b = (__bf16*)(ws + OFF_WFC2);
  float*  atnp  = (float*) (ws + OFF_ATNP);

  // TDM probe (architectural no-op: NULL descriptor)
  tdm_probe_kernel<<<1, 32, 0, stream>>>();

  // weight conversions (tiny)
  cvt_bf16_kernel<<<(576 * 192 + 255) / 256, 256, 0, stream>>>(w_qkv, wqkvb, 576 * 192);
  cvt_bf16_kernel<<<(192 * 192 + 255) / 256, 256, 0, stream>>>(w_proj, wprjb, 192 * 192);
  cvt_bf16_kernel<<<(384 * 192 + 255) / 256, 256, 0, stream>>>(w_fc1, wfc1b, 384 * 192);
  cvt_bf16_kernel<<<(192 * 384 + 255) / 256, 256, 0, stream>>>(w_fc2, wfc2b, 192 * 384);

  // 1) LayerNorm1
  ln1_kernel<<<NPIX, DIM, 0, stream>>>(x, g1, b1, x1f, x1b);

  // 2) qkv 1x1 GEMM: 4096 pixel-tiles * 18 oc-pair-tiles = 73728 waves
  qkv_gemm_kernel<<<(NPIX / 16) * (C3 / 32) / 8, 256, 0, stream>>>(x1b, wqkvb, qkvp);

  // 3) depthwise 3x3
  dwconv_kernel<<<(BATCH * C3 * HW + 255) / 256, 256, 0, stream>>>(qkvp, w_dw, qkvf);

  // 4) l2norm q,k over hw + bf16 conversion of q,k,v
  qk_norm_cvt_kernel<<<BATCH * C3, 256, 0, stream>>>(qkvf, qkvb);

  // 5) attn = q k^T partials (24 bh * 4 tiles * 16 K-chunks = 1536 waves) + reduce*temp
  attn_qk_part_kernel<<<1536 / 8, 256, 0, stream>>>(qkvb, atnp);
  attn_reduce_kernel<<<(BATCH * HEADS * CP * CP + 255) / 256, 256, 0, stream>>>(atnp, temp, atnr);

  // 6) head conv + softmax
  attn_head_conv_kernel<<<(BATCH * HEADS * CP * CP + 255) / 256, 256, 0, stream>>>(atnr, w_head, atnc);
  softmax_kernel<<<BATCH * HEADS * CP, 32, 0, stream>>>(atnc, atnb);

  // 7) out = attn @ v : 49152 waves
  attn_v_kernel<<<49152 / 8, 256, 0, stream>>>(atnb, qkvb, outb);

  // 8) proj 1x1 + residual -> x2 pixel-major : 4096 * 6 = 24576 waves
  proj_gemm_kernel<<<(NPIX / 16) * (DIM / 32) / 8, 256, 0, stream>>>(outb, wprjb, x1f, x2);

  // 9) LayerNorm2
  ln2_kernel<<<NPIX, DIM, 0, stream>>>(x2, g2, b2, ybf);

  // 10) fc1 + GELU : 4096 * 12 = 49152 waves
  fc1_gemm_kernel<<<(NPIX / 16) * (HIDDEN / 32) / 8, 256, 0, stream>>>(ybf, wfc1b, b_fc1, h1);

  // 11) fc2 + GELU -> y4 channel-major : 24576 waves
  fc2_gemm_kernel<<<(NPIX / 16) * (DIM / 32) / 8, 256, 0, stream>>>(h1, wfc2b, b_fc2, y4);

  // 12) spatial gate
  gate_kernel<<<(BATCH * HW + 255) / 256, 256, 0, stream>>>(y4, w_sa, b_sa, gate);

  // 13) gate-multiply + residual + transpose to NCHW
  final_kernel<<<(BATCH * DIM * HW + 255) / 256, 256, 0, stream>>>(x2, y4, gate, out);
}
